// VoxelAttention_29257317220739
// MI455X (gfx1250) — matrix operand order repaired and checked by reading, not compiled, not run on hardware
//
#include <hip/hip_runtime.h>
#include <hip/hip_bf16.h>

#define DEV __device__ __forceinline__

typedef __attribute__((ext_vector_type(16))) __bf16 v16bf;
typedef __attribute__((ext_vector_type(8)))  __bf16 v8bf;
typedef __attribute__((ext_vector_type(8)))  float  v8f;

constexpr int BB = 2, NN = 16384, MM = 2048, DD = 256;
constexpr int NSPLIT = 8;                    // n-loop split for k_attn_partial
constexpr int NCHUNK = NN / NSPLIT;          // 2048 columns per wave
constexpr float SCALE  = 0.0625f;            // 1/sqrt(256)
constexpr float LN_EPS = 1e-5f;

DEV v8f wmma_bf16(v16bf a, v16bf b, v8f c) {
  return __builtin_amdgcn_wmma_f32_16x16x32_bf16(false, a, false, b, (short)0, c,
                                                 false, false);
}

DEV v8f zero8() { v8f z = {0.f,0.f,0.f,0.f,0.f,0.f,0.f,0.f}; return z; }

// A operand (16x32, M x K, bf16) from a row-major f32 tile. ld in elements.
// Lane L (<16): rows M=L, K groups {0..7, 16..23}; lane>=16: K {8..15, 24..31}.
DEV v16bf load_a_f32(const float* p, int ld) {
  int lane = threadIdx.x & 31, r = lane & 15, hi = lane >> 4;
  const float* p0 = p + (size_t)r * ld + hi * 8;
  v16bf a;
#pragma unroll
  for (int i = 0; i < 8; ++i) a[i] = (__bf16)p0[i];
#pragma unroll
  for (int i = 0; i < 8; ++i) a[8 + i] = (__bf16)p0[16 + i];
  return a;
}

// A operand from a row-major bf16 tile.
DEV v16bf load_a_bf16(const __bf16* p, int ld) {
  int lane = threadIdx.x & 31, r = lane & 15, hi = lane >> 4;
  const __bf16* p0 = p + (size_t)r * ld + hi * 8;
  union { v16bf v; v8bf h[2]; } u;
  u.h[0] = *(const v8bf*)p0;
  u.h[1] = *(const v8bf*)(p0 + 16);
  return u.v;
}

// B operand (32x16, K x N, bf16) from S[col][k] row-major (i.e. B pre-transposed).
// Lane L (<16): column N=L, K groups {0..7, 16..23}; lane>=16: column N=L-16, K {8..15, 24..31}.
DEV v16bf load_b_bf16(const __bf16* p, int ld) {
  int lane = threadIdx.x & 31, c = lane & 15, hi = lane >> 4;
  const __bf16* p0 = p + (size_t)c * ld + hi * 8;
  union { v16bf v; v8bf h[2]; } u;
  u.h[0] = *(const v8bf*)p0;
  u.h[1] = *(const v8bf*)(p0 + 16);
  return u.v;
}

// ---------------- K0: positional MLP front-end: rel[b,n,3] -------------------
__global__ void k_rel(const float* __restrict__ p_xyz, const float* __restrict__ v_xyz,
                      const float* __restrict__ Wp1, const float* __restrict__ bp1,
                      const float* __restrict__ ln_w, const float* __restrict__ ln_b,
                      float* __restrict__ rel) {
  int i = blockIdx.x * blockDim.x + threadIdx.x;   // flat b*N + n, exactly B*N threads
  int b = i >> 14;                                 // N = 16384 = 2^14
  float d0 = fabsf(p_xyz[i * 3 + 0] - v_xyz[b * 3 + 0]);
  float d1 = fabsf(p_xyz[i * 3 + 1] - v_xyz[b * 3 + 1]);
  float d2 = fabsf(p_xyz[i * 3 + 2] - v_xyz[b * 3 + 2]);
  float h0 = d0 * Wp1[0] + d1 * Wp1[3] + d2 * Wp1[6] + bp1[0];
  float h1 = d0 * Wp1[1] + d1 * Wp1[4] + d2 * Wp1[7] + bp1[1];
  float h2 = d0 * Wp1[2] + d1 * Wp1[5] + d2 * Wp1[8] + bp1[2];
  float mu = (h0 + h1 + h2) * (1.0f / 3.0f);
  float e0 = h0 - mu, e1 = h1 - mu, e2 = h2 - mu;
  float var = (e0 * e0 + e1 * e1 + e2 * e2) * (1.0f / 3.0f);
  float inv = rsqrtf(var + LN_EPS);
  rel[i * 3 + 0] = fmaxf(e0 * inv * ln_w[0] + ln_b[0], 0.f);
  rel[i * 3 + 1] = fmaxf(e1 * inv * ln_w[1] + ln_b[1], 0.f);
  rel[i * 3 + 2] = fmaxf(e2 * inv * ln_w[2] + ln_b[2], 0.f);
}

// ---------------- K_wt: transpose+convert Wq/Wk/Wv -> bf16 [Dout][Din] -------
__global__ void k_wt(const float* __restrict__ Wq, const float* __restrict__ Wk,
                     const float* __restrict__ Wv,
                     __bf16* __restrict__ WqT, __bf16* __restrict__ WkT,
                     __bf16* __restrict__ WvT) {
  int i = blockIdx.x * blockDim.x + threadIdx.x;   // 3 * D*D threads
  int which = i >> 16;                              // D*D = 65536
  int j = i & 65535;
  int k = j >> 8, d = j & 255;
  const float* W = which == 0 ? Wq : (which == 1 ? Wk : Wv);
  __bf16*      T = which == 0 ? WqT : (which == 1 ? WkT : WvT);
  T[(size_t)d * DD + k] = (__bf16)W[(size_t)k * DD + d];
}

// ---------------- K1: kpv = pf@Wk + bk + rel*Wp2 + bp2 ; vT = (pf@Wv + bv)^T --
__global__ void __launch_bounds__(128) k_proj_kv(
    const float* __restrict__ pf, const __bf16* __restrict__ WkT,
    const __bf16* __restrict__ WvT, const float* __restrict__ bk,
    const float* __restrict__ bv, const float* __restrict__ bp2,
    const float* __restrict__ Wp2, const float* __restrict__ rel,
    __bf16* __restrict__ kpv, __bf16* __restrict__ vT) {
  int wave = threadIdx.x >> 5;
  int tile = blockIdx.x * 4 + wave;      // (B*N/16) * (D/16) = 32768 tiles
  int rt = tile >> 4;                    // flat row-tile over B*N
  int dt = tile & 15;
  int lane = threadIdx.x & 31, c = lane & 15, hi = lane >> 4;
  int row0 = rt * 16;
  v8f ck = zero8(), cv = zero8();
#pragma unroll
  for (int t = 0; t < 8; ++t) {
    int kk = t * 32;
    v16bf a  = load_a_f32(pf + (size_t)row0 * DD + kk, DD);
    v16bf b1 = load_b_bf16(WkT + (size_t)(dt * 16) * DD + kk, DD);
    v16bf b2 = load_b_bf16(WvT + (size_t)(dt * 16) * DD + kk, DD);
    ck = wmma_bf16(a, b1, ck);
    cv = wmma_bf16(a, b2, cv);
  }
  int d = dt * 16 + c;
  float addk = bk[d] + bp2[d];
  float w0 = Wp2[d], w1 = Wp2[DD + d], w2 = Wp2[2 * DD + d];
  float bvd = bv[d];
#pragma unroll
  for (int r = 0; r < 8; ++r) {
    int row = row0 + r + 8 * hi;         // flat b*N + n
    const float* rp = rel + (size_t)row * 3;
    float pv = rp[0] * w0 + rp[1] * w1 + rp[2] * w2;
    kpv[(size_t)row * DD + d] = (__bf16)(ck[r] + addk + pv);
    int bb = row >> 14, n = row & (NN - 1);
    vT[((size_t)bb * DD + d) * NN + n] = (__bf16)(cv[r] + bvd);
  }
}

// ---------------- K2: q = vf@Wq + bq  (bf16 row-major) -----------------------
__global__ void __launch_bounds__(128) k_proj_q(
    const float* __restrict__ vf, const __bf16* __restrict__ WqT,
    const float* __restrict__ bq, __bf16* __restrict__ qb) {
  int wave = threadIdx.x >> 5;
  int tile = blockIdx.x * 4 + wave;      // (B*M/16) * (D/16) = 4096 tiles
  int rt = tile >> 4;
  int dt = tile & 15;
  int lane = threadIdx.x & 31, c = lane & 15, hi = lane >> 4;
  int row0 = rt * 16;
  v8f cq = zero8();
#pragma unroll
  for (int t = 0; t < 8; ++t) {
    int kk = t * 32;
    v16bf a = load_a_f32(vf + (size_t)row0 * DD + kk, DD);
    v16bf b = load_b_bf16(WqT + (size_t)(dt * 16) * DD + kk, DD);
    cq = wmma_bf16(a, b, cq);
  }
  int d = dt * 16 + c;
  float bqd = bq[d];
#pragma unroll
  for (int r = 0; r < 8; ++r) {
    int row = row0 + r + 8 * hi;
    qb[(size_t)row * DD + d] = (__bf16)(cq[r] + bqd);
  }
}

// ---------------- K3: per-column (n) softmax stats over m --------------------
__global__ void __launch_bounds__(128) k_stats(
    const __bf16* __restrict__ qb, const __bf16* __restrict__ kpv,
    float* __restrict__ colmax, float* __restrict__ colsum) {
  int wave = threadIdx.x >> 5;
  int tile = blockIdx.x * 4 + wave;      // B * (N/16) = 2048 tiles
  int b = tile >> 10;
  int n0 = (tile & 1023) * 16;
  int lane = threadIdx.x & 31;
  v16bf Bt[8];
#pragma unroll
  for (int t = 0; t < 8; ++t)
    Bt[t] = load_b_bf16(kpv + ((size_t)b * NN + n0) * DD + t * 32, DD);
  float run_max = -3e38f, run_sum = 0.f;
  for (int m0 = 0; m0 < MM; m0 += 16) {
    v8f cacc = zero8();
#pragma unroll
    for (int t = 0; t < 8; ++t) {
      v16bf a = load_a_bf16(qb + ((size_t)b * MM + m0) * DD + t * 32, DD);
      cacc = wmma_bf16(a, Bt[t], cacc);
    }
    float tmax = -3e38f;
#pragma unroll
    for (int r = 0; r < 8; ++r) tmax = fmaxf(tmax, cacc[r]);
    tmax = fmaxf(tmax, __shfl_xor(tmax, 16, 32));   // join rows 0-7 with 8-15
    float nm = fmaxf(run_max, tmax * SCALE);
    float s = 0.f;
#pragma unroll
    for (int r = 0; r < 8; ++r) s += __expf(cacc[r] * SCALE - nm);
    s += __shfl_xor(s, 16, 32);
    run_sum = run_sum * __expf(run_max - nm) + s;
    run_max = nm;
  }
  if (lane < 16) {
    colmax[(size_t)b * NN + n0 + lane] = run_max;
    colsum[(size_t)b * NN + n0 + lane] = run_sum;
  }
}

// ------- K4: partial res over an n-chunk; logits computed once per (mt,chunk) -
// Wave owns (b, m-tile 16, n-chunk 2048) with FULL D=256 accumulation:
// 16 logits WMMAs + 16 accumulation WMMAs per 32-column step, zero d-redundancy.
__global__ void __launch_bounds__(256) k_attn_partial(
    const __bf16* __restrict__ qb, const __bf16* __restrict__ kpv,
    const __bf16* __restrict__ vT, const float* __restrict__ colmax,
    const float* __restrict__ colsum, float* __restrict__ partial) {
  __shared__ __bf16 lds[8][16 * 32];     // per-wave 16(m) x 32(n) staging tile
  int wave = threadIdx.x >> 5;
  int lane = threadIdx.x & 31, c = lane & 15, hi = lane >> 4;
  int tile = blockIdx.x * 8 + wave;      // B * (M/16) * NSPLIT = 2048 tiles
  int b   = tile >> 10;                  // (M/16)*NSPLIT = 1024
  int rem = tile & 1023;
  int mt  = rem >> 3;                    // 0..127
  int ns  = rem & 7;                     // 0..7
  int m0 = mt * 16;
  int nbase = ns * NCHUNK;
  __bf16* wl = &lds[wave][0];

  v16bf Aq[8];                            // q A-tiles hoisted over the n loop
#pragma unroll
  for (int t = 0; t < 8; ++t)
    Aq[t] = load_a_bf16(qb + ((size_t)b * MM + m0) * DD + t * 32, DD);

  v8f acc[16];
#pragma unroll
  for (int t = 0; t < 16; ++t) acc[t] = zero8();

  for (int n0 = nbase; n0 < nbase + NCHUNK; n0 += 32) {
#pragma unroll
    for (int sub = 0; sub < 2; ++sub) {
      int nn = n0 + sub * 16;
      v8f lc = zero8();
#pragma unroll
      for (int t = 0; t < 8; ++t) {
        v16bf bb = load_b_bf16(kpv + ((size_t)b * NN + nn) * DD + t * 32, DD);
        lc = wmma_bf16(Aq[t], bb, lc);
      }
      float cm  = colmax[(size_t)b * NN + nn + c];
      float inv = 1.f / colsum[(size_t)b * NN + nn + c];
#pragma unroll
      for (int r = 0; r < 8; ++r) {
        float wv = __expf(lc[r] * SCALE - cm) * inv;
        wl[(r + 8 * hi) * 32 + sub * 16 + c] = (__bf16)wv;
      }
    }
    asm volatile("s_wait_dscnt 0" ::: "memory");   // LDS stores -> loads ordering
    v16bf Aw;
    {
      union { v16bf v; v8bf h[2]; } u;
      u.h[0] = *(const v8bf*)&wl[c * 32 + hi * 8];
      u.h[1] = *(const v8bf*)&wl[c * 32 + 16 + hi * 8];
      Aw = u.v;
    }
#pragma unroll
    for (int t = 0; t < 16; ++t) {
      v16bf vb = load_b_bf16(vT + ((size_t)b * DD + t * 16) * NN + n0, NN);
      acc[t] = wmma_bf16(Aw, vb, acc[t]);
    }
  }
  float* pout = partial + (size_t)ns * ((size_t)BB * MM * DD);
#pragma unroll
  for (int t = 0; t < 16; ++t) {
    int d = t * 16 + c;
#pragma unroll
    for (int r = 0; r < 8; ++r) {
      size_t idx = ((size_t)b * MM + m0 + r + 8 * hi) * (size_t)DD + d;
      pout[idx] = acc[t][r];
    }
  }
}

// ---------------- K5: deterministic reduction of n-chunk partials + residual -
__global__ void k_reduce(const float* __restrict__ partial,
                         const float* __restrict__ vf, float* __restrict__ out) {
  size_t i = (size_t)blockIdx.x * blockDim.x + threadIdx.x;  // B*M*D threads
  constexpr size_t STRIDE = (size_t)BB * MM * DD;
  float s = vf[i];
#pragma unroll
  for (int ns = 0; ns < NSPLIT; ++ns) s += partial[(size_t)ns * STRIDE + i];
  out[i] = s;
}

extern "C" void kernel_launch(void* const* d_in, const int* in_sizes, int n_in,
                              void* d_out, int out_size, void* d_ws, size_t ws_size,
                              hipStream_t stream) {
  const float* p_xyz = (const float*)d_in[0];
  const float* v_xyz = (const float*)d_in[1];
  const float* p_features = (const float*)d_in[2];
  const float* v_features = (const float*)d_in[3];
  const float* Wq  = (const float*)d_in[4];
  const float* bq  = (const float*)d_in[5];
  const float* Wk  = (const float*)d_in[6];
  const float* bk  = (const float*)d_in[7];
  const float* Wv  = (const float*)d_in[8];
  const float* bv  = (const float*)d_in[9];
  const float* Wp1 = (const float*)d_in[10];
  const float* bp1 = (const float*)d_in[11];
  const float* ln_w = (const float*)d_in[12];
  const float* ln_b = (const float*)d_in[13];
  const float* Wp2 = (const float*)d_in[14];
  const float* bp2 = (const float*)d_in[15];
  float* out = (float*)d_out;

  char* ws = (char*)d_ws;
  size_t off = 0;
  auto alloc = [&](size_t bytes) -> void* {
    void* p = ws + off;
    off += (bytes + 255) & ~(size_t)255;
    return p;
  };
  float*  rel     = (float*)alloc((size_t)BB * NN * 3 * sizeof(float));
  __bf16* WqT     = (__bf16*)alloc((size_t)DD * DD * sizeof(__bf16));
  __bf16* WkT     = (__bf16*)alloc((size_t)DD * DD * sizeof(__bf16));
  __bf16* WvT     = (__bf16*)alloc((size_t)DD * DD * sizeof(__bf16));
  __bf16* qb      = (__bf16*)alloc((size_t)BB * MM * DD * sizeof(__bf16));
  __bf16* kpv     = (__bf16*)alloc((size_t)BB * NN * DD * sizeof(__bf16));
  __bf16* vT      = (__bf16*)alloc((size_t)BB * DD * NN * sizeof(__bf16));
  float*  colmax  = (float*)alloc((size_t)BB * NN * sizeof(float));
  float*  colsum  = (float*)alloc((size_t)BB * NN * sizeof(float));
  float*  partial = (float*)alloc((size_t)NSPLIT * BB * MM * DD * sizeof(float));
  (void)in_sizes; (void)n_in; (void)out_size; (void)ws_size;

  k_rel<<<(BB * NN) / 256, 256, 0, stream>>>(p_xyz, v_xyz, Wp1, bp1, ln_w, ln_b, rel);
  k_wt<<<(3 * DD * DD) / 256, 256, 0, stream>>>(Wq, Wk, Wv, WqT, WkT, WvT);
  k_proj_kv<<<((BB * NN / 16) * (DD / 16)) / 4, 128, 0, stream>>>(
      p_features, WkT, WvT, bk, bv, bp2, Wp2, rel, kpv, vT);
  k_proj_q<<<((BB * MM / 16) * (DD / 16)) / 4, 128, 0, stream>>>(
      v_features, WqT, bq, qb);
  k_stats<<<(BB * (NN / 16)) / 4, 128, 0, stream>>>(qb, kpv, colmax, colsum);
  k_attn_partial<<<(BB * (MM / 16) * NSPLIT) / 8, 256, 0, stream>>>(
      qb, kpv, vT, colmax, colsum, partial);
  k_reduce<<<(BB * MM * DD) / 256, 256, 0, stream>>>(partial, v_features, out);
}
